// FastAttention_62560493633840
// MI455X (gfx1250) — matrix operand order repaired and checked by reading, not compiled, not run on hardware
//
#include <hip/hip_runtime.h>

typedef _Float16 v16h __attribute__((ext_vector_type(16)));
typedef _Float16 h8   __attribute__((ext_vector_type(8)));
typedef _Float16 h4   __attribute__((ext_vector_type(4)));
typedef _Float16 h2   __attribute__((ext_vector_type(2)));
typedef float    v8f  __attribute__((ext_vector_type(8)));

#define BB 4
#define NN 2048
#define HH 8
#define DD 64
#define BK 64
#define NT (NN / BK)          // 32 K/V tiles
#define KT_STR 72             // halfs per K-tile row  (144B rows, 16B aligned)
#define VT_STR 72             // halfs per V^T row     (144B rows, 16B aligned)
#define P_STR  72             // halfs per P row
// (1/sqrt(64)) * log2(e): fold head scale + exp->exp2 conversion into Q
#define QSCALE 0.1803368801111f

static __device__ __forceinline__ v16h cat16(h8 lo, h8 hi) {
    return __builtin_shufflevector(lo, hi,
        0,1,2,3,4,5,6,7,8,9,10,11,12,13,14,15);
}

// one v_cvt_pk_rtz_f16_f32 per float pair (builtin returns __fp16x2; element
// copies are no-ops at ISA level but keep C++ vector types consistent)
static __device__ __forceinline__ h2 pk2(float a, float b) {
    auto t = __builtin_amdgcn_cvt_pkrtz(a, b);
    h2 r; r[0] = (_Float16)t[0]; r[1] = (_Float16)t[1];
    return r;
}
static __device__ __forceinline__ h8 mk8(h2 a, h2 b, h2 c, h2 d) {
    h4 lo = __builtin_shufflevector(a, b, 0, 1, 2, 3);
    h4 hi = __builtin_shufflevector(c, d, 0, 1, 2, 3);
    return __builtin_shufflevector(lo, hi, 0, 1, 2, 3, 4, 5, 6, 7);
}

// xor-butterfly within each 16-lane half via v_permlane16_b32 (VALU, not DS)
static __device__ __forceinline__ float plx(float x, unsigned s0, unsigned s1) {
    return __int_as_float(__builtin_amdgcn_permlane16(
        __float_as_int(x), __float_as_int(x), s0, s1, false, false));
}
static __device__ __forceinline__ float red_max16(float t) {
    t = fmaxf(t, plx(t, 0x67452301u, 0xEFCDAB89u));  // xor 1
    t = fmaxf(t, plx(t, 0x54761032u, 0xDCFE98BAu));  // xor 2
    t = fmaxf(t, plx(t, 0x32107654u, 0xBA98FEDCu));  // xor 4
    t = fmaxf(t, plx(t, 0xFEDCBA98u, 0x76543210u));  // xor 8
    return t;
}
static __device__ __forceinline__ float red_sum16(float t) {
    t += plx(t, 0x67452301u, 0xEFCDAB89u);
    t += plx(t, 0x54761032u, 0xDCFE98BAu);
    t += plx(t, 0x32107654u, 0xBA98FEDCu);
    t += plx(t, 0xFEDCBA98u, 0x76543210u);
    return t;
}

__global__ __launch_bounds__(256, 1)
void fa_fwd_kernel(const float* __restrict__ q,
                   const float* __restrict__ k,
                   const float* __restrict__ v,
                   float* __restrict__ out)
{
    __shared__ _Float16 sK[2][BK * KT_STR];    // K tile,  row-major [kv][d]
    __shared__ _Float16 sVT[2][DD * VT_STR];   // V tile transposed  [d][kv]
    __shared__ _Float16 sP[8][16 * P_STR];     // per-wave P staging

    const int tid  = threadIdx.x;
    const int wave = tid >> 5;
    const int lane = tid & 31;
    const int lh   = lane & 15;
    const int hs   = lane >> 4;

    const int bh = blockIdx.y;
    const int bb = bh >> 3;
    const int hh = bh & 7;
    const int m0 = blockIdx.x * 128 + wave * 16;

    // ---- Q into two A-fragments (d 0..31 / 32..63), scale folded in ----
    v16h aq0, aq1;
    {
        const float* qrow = q + (((size_t)bb * NN + (m0 + lh)) * HH + hh) * DD;
        const int d0 = hs * 8;
        float qs[32];
#pragma unroll
        for (int i = 0; i < 8; ++i) {
            qs[i]      = qrow[d0 + i]           * QSCALE;
            qs[8 + i]  = qrow[d0 + 16 + i]      * QSCALE;
            qs[16 + i] = qrow[32 + d0 + i]      * QSCALE;
            qs[24 + i] = qrow[32 + d0 + 16 + i] * QSCALE;
        }
        aq0 = cat16(mk8(pk2(qs[0],  qs[1]),  pk2(qs[2],  qs[3]),
                        pk2(qs[4],  qs[5]),  pk2(qs[6],  qs[7])),
                    mk8(pk2(qs[8],  qs[9]),  pk2(qs[10], qs[11]),
                        pk2(qs[12], qs[13]), pk2(qs[14], qs[15])));
        aq1 = cat16(mk8(pk2(qs[16], qs[17]), pk2(qs[18], qs[19]),
                        pk2(qs[20], qs[21]), pk2(qs[22], qs[23])),
                    mk8(pk2(qs[24], qs[25]), pk2(qs[26], qs[27]),
                        pk2(qs[28], qs[29]), pk2(qs[30], qs[31])));
    }

    v8f o[4] = {};
    float mrun[8], lrun[8];
#pragma unroll
    for (int j = 0; j < 8; ++j) { mrun[j] = -__builtin_inff(); lrun[j] = 0.f; }

    _Float16* sPw = &sP[wave][0];

    // loader mappings (256 threads cover a 64x64 tile twice: K and V)
    const int rk = tid >> 2;            // K row 0..63
    const int ck = (tid & 3) * 16;      // K col 0,16,32,48
    const int rv = (tid >> 4) * 4;      // V rows r..r+3
    const int cv = (tid & 15) * 4;      // V cols c..c+3
    const float* kbase = k + (((size_t)bb * NN) * HH + hh) * DD;
    const float* vbase = v + (((size_t)bb * NN) * HH + hh) * DD;

    float kreg[16], vreg[16];

#define LOAD_TILE_REGS(KT)                                                    \
    {                                                                         \
        const float* kr = kbase + (size_t)((KT) * BK + rk) * (HH * DD) + ck;  \
        _Pragma("unroll")                                                     \
        for (int i = 0; i < 16; ++i) kreg[i] = kr[i];                         \
        _Pragma("unroll")                                                     \
        for (int i = 0; i < 4; ++i) {                                         \
            const float* vr =                                                 \
                vbase + (size_t)((KT) * BK + rv + i) * (HH * DD) + cv;        \
            _Pragma("unroll")                                                 \
            for (int j2 = 0; j2 < 4; ++j2) vreg[i * 4 + j2] = vr[j2];         \
        }                                                                     \
    }

#define STAGE_TILE(BUF)                                                       \
    {                                                                         \
        h8 t0 = mk8(pk2(kreg[0],  kreg[1]),  pk2(kreg[2],  kreg[3]),          \
                    pk2(kreg[4],  kreg[5]),  pk2(kreg[6],  kreg[7]));         \
        h8 t1 = mk8(pk2(kreg[8],  kreg[9]),  pk2(kreg[10], kreg[11]),         \
                    pk2(kreg[12], kreg[13]), pk2(kreg[14], kreg[15]));        \
        *(h8*)&sK[BUF][rk * KT_STR + ck]     = t0;                            \
        *(h8*)&sK[BUF][rk * KT_STR + ck + 8] = t1;                            \
        _Pragma("unroll")                                                     \
        for (int j2 = 0; j2 < 4; ++j2) {                                      \
            h4 p = __builtin_shufflevector(                                   \
                pk2(vreg[j2], vreg[4 + j2]),                                  \
                pk2(vreg[8 + j2], vreg[12 + j2]), 0, 1, 2, 3);                \
            *(h4*)&sVT[BUF][(cv + j2) * VT_STR + rv] = p;                     \
        }                                                                     \
    }

    // ---- prologue: fetch + stage tile 0 ----
    LOAD_TILE_REGS(0)
    STAGE_TILE(0)
    __syncthreads();

    for (int kt = 0; kt < NT; ++kt) {
        const int cur = kt & 1, nxt = cur ^ 1;
        const bool has_next = (kt + 1) < NT;

        // ---- issue global loads for tile kt+1 early (hide under WMMA) ----
        if (has_next) LOAD_TILE_REGS(kt + 1)

        // ---- S = (Q*scale) K^T : 4 N-subtiles x 2 K-steps ----
        v8f s[4];
#pragma unroll
        for (int ns = 0; ns < 4; ++ns) {
            v8f acc = {};
#pragma unroll
            for (int ks = 0; ks < 2; ++ks) {
                const v16h a = ks ? aq1 : aq0;
                const _Float16* kb =
                    &sK[cur][(ns * 16 + lh) * KT_STR + ks * 32 + hs * 8];
                v16h b = cat16(*(const h8*)kb, *(const h8*)(kb + 16));
                acc = __builtin_amdgcn_wmma_f32_16x16x32_f16(
                          false, a, false, b, (short)0, acc, false, false);
            }
            s[ns] = acc;
        }

        // ---- online softmax over 64 columns ----
        float scl[8];
#pragma unroll
        for (int j = 0; j < 8; ++j) {
            float t = fmaxf(fmaxf(s[0][j], s[1][j]), fmaxf(s[2][j], s[3][j]));
            t = red_max16(t);
            const float mn = fmaxf(mrun[j], t);
            scl[j] = __builtin_amdgcn_exp2f(mrun[j] - mn);
            mrun[j] = mn;
        }
#pragma unroll
        for (int j = 0; j < 8; ++j) {
            float p0 = __builtin_amdgcn_exp2f(s[0][j] - mrun[j]);
            float p1 = __builtin_amdgcn_exp2f(s[1][j] - mrun[j]);
            float p2 = __builtin_amdgcn_exp2f(s[2][j] - mrun[j]);
            float p3 = __builtin_amdgcn_exp2f(s[3][j] - mrun[j]);
            float t = red_sum16((p0 + p1) + (p2 + p3));
            lrun[j] = lrun[j] * scl[j] + t;
            // pack P pairs: one pkrtz feeds a b16 store (lo) + b16_d16_hi store
            h2 pa = pk2(p0, p1);
            h2 pb2 = pk2(p2, p3);
            const int prow = (j + 8 * hs) * P_STR;
            sPw[prow + lh]      = pa[0];
            sPw[prow + 16 + lh] = pa[1];
            sPw[prow + 32 + lh] = pb2[0];
            sPw[prow + 48 + lh] = pb2[1];
        }
#pragma unroll
        for (int j = 0; j < 8; ++j) {
            o[0][j] *= scl[j]; o[1][j] *= scl[j];
            o[2][j] *= scl[j]; o[3][j] *= scl[j];
        }
        asm volatile("s_wait_dscnt 0" ::: "memory");

        // re-read P as two A-fragments (kv 0..31 / 32..63)
        const _Float16* pb = &sPw[lh * P_STR + hs * 8];
        v16h ap0 = cat16(*(const h8*)pb,        *(const h8*)(pb + 16));
        v16h ap1 = cat16(*(const h8*)(pb + 32), *(const h8*)(pb + 48));

        // ---- O += P V : 4 d-subtiles x 2 K-steps ----
#pragma unroll
        for (int ds = 0; ds < 4; ++ds) {
            const _Float16* vb0 = &sVT[cur][(ds * 16 + lh) * VT_STR + hs * 8];
            v16h bv0 = cat16(*(const h8*)vb0, *(const h8*)(vb0 + 16));
            o[ds] = __builtin_amdgcn_wmma_f32_16x16x32_f16(
                        false, ap0, false, bv0, (short)0, o[ds], false, false);
            const _Float16* vb1 = vb0 + 32;
            v16h bv1 = cat16(*(const h8*)vb1, *(const h8*)(vb1 + 16));
            o[ds] = __builtin_amdgcn_wmma_f32_16x16x32_f16(
                        false, ap1, false, bv1, (short)0, o[ds], false, false);
        }

        // ---- convert + stage tile kt+1 into the other buffer ----
        if (has_next) STAGE_TILE(nxt)
        __syncthreads();
    }

    // ---- normalize and store [b, n, h, d] ----
    float inv[8];
#pragma unroll
    for (int j = 0; j < 8; ++j) inv[j] = 1.0f / lrun[j];
#pragma unroll
    for (int ds = 0; ds < 4; ++ds) {
#pragma unroll
        for (int j = 0; j < 8; ++j) {
            const int m = m0 + j + 8 * hs;
            out[(((size_t)bb * NN + m) * HH + hh) * DD + ds * 16 + lh] =
                o[ds][j] * inv[j];
        }
    }
}

extern "C" void kernel_launch(void* const* d_in, const int* in_sizes, int n_in,
                              void* d_out, int out_size, void* d_ws, size_t ws_size,
                              hipStream_t stream) {
    (void)in_sizes; (void)n_in; (void)out_size; (void)d_ws; (void)ws_size;
    const float* q = (const float*)d_in[0];
    const float* k = (const float*)d_in[1];
    const float* v = (const float*)d_in[2];
    float* out = (float*)d_out;
    dim3 grid(NN / 128, BB * HH);
    dim3 block(256);
    fa_fwd_kernel<<<grid, block, 0, stream>>>(q, k, v, out);
}
